// Untied_LeNet_55645596287383
// MI455X (gfx1250) — matrix operand order repaired and checked by reading, not compiled, not run on hardware
//
#include <hip/hip_runtime.h>

typedef __attribute__((ext_vector_type(16))) _Float16 v16h;
typedef __attribute__((ext_vector_type(8)))  _Float16 v8h;
typedef __attribute__((ext_vector_type(8)))  float    v8f;
typedef __attribute__((ext_vector_type(4)))  int      v4i;

union AFrag { v16h v; v8h h[2]; };

__device__ __forceinline__ v8f wmma_f16(v16h a, v16h b, v8f c) {
  // (neg_a, A, neg_b, B, c_mod, C, reuse_a, reuse_b)
  return __builtin_amdgcn_wmma_f32_16x16x32_f16(false, a, false, b, (short)0, c,
                                                false, false);
}

// ---- gfx1250 async global->LDS copy (ASYNCcnt path), guarded ----
#if defined(__has_builtin)
#if __has_builtin(__builtin_amdgcn_global_load_async_to_lds_b128)
#define HAVE_ASYNC_LDS 1
#endif
#endif

#ifdef HAVE_ASYNC_LDS
typedef __attribute__((address_space(1))) v4i GV4;  // global (prints as __device__)
typedef __attribute__((address_space(3))) v4i LV4;  // LDS    (prints as __shared__)
__device__ __forceinline__ void async_g2l_b128(const void* g, void* l) {
  __builtin_amdgcn_global_load_async_to_lds_b128((GV4*)g, (LV4*)l, 0, 0);
}
__device__ __forceinline__ void wait_async0() {
#if __has_builtin(__builtin_amdgcn_s_wait_asynccnt)
  __builtin_amdgcn_s_wait_asynccnt(0);
#else
  asm volatile("s_wait_asynccnt 0x0" ::: "memory");
#endif
}
#endif

// ---------------------------------------------------------------------------
// Untied (locally-connected) 5x5 conv + bias + ReLU as per-site WMMA GEMMs.
// XT = input element type (f32 for layer 1, f16 afterwards); output f16.
// 128 threads = 4 wave32s; each wave owns 32 batch rows (2 M-tiles / 2 WMMAs
// per K chunk, shared B fragment). K-chunk loop fully unrolled (KC = 3 / 5).
// A loads are batched into registers first so they issue back-to-back.
// ---------------------------------------------------------------------------
template <typename XT, int Cin, int Hin, int Win, int Oc, int Ho, int Wo>
__global__ __launch_bounds__(128) void localconv_wmma(
    const XT* __restrict__ X, const float* __restrict__ W,
    const float* __restrict__ Bias, _Float16* __restrict__ Out)
{
  constexpr int K   = Cin * 25;
  constexpr int KC  = (K + 31) / 32;
  constexpr int CHW = Cin * Hin * Win;     // batch stride of X
  constexpr int OHW = Oc * Ho * Wo;        // batch stride of Out
  constexpr int WN  = Cin * Ho * Wo * 25;  // out-channel stride of W

  __shared__ __align__(32) _Float16 As[4][32][32];
  __shared__ __align__(32) _Float16 Bs[16][32];

  const int tid   = threadIdx.x;
  const int lane  = tid & 31;
  const int wv    = tid >> 5;
  const int site  = blockIdx.x;
  const int ho    = site / Wo;
  const int wo    = site - ho * Wo;
  const int mBase = blockIdx.y * 128 + wv * 32;

  v8f acc0 = {}, acc1 = {};

#pragma unroll
  for (int kc = 0; kc < KC; ++kc) {
    // lane <-> k column; decode (c,kh,kw) once per chunk
    const int  kg     = kc * 32 + lane;
    const bool kvalid = (kg < K);
    const int  kgc    = kvalid ? kg : (K - 1);   // clamped: garbage*0 = 0
    const int  c      = kgc / 25;
    const int  r      = kgc - c * 25;
    const int  kh     = r / 5;
    const int  kw     = r - kh * 5;

    // ---- A: batch 32 loads into regs, then convert+store to LDS ----
    const XT* xp = X + (long)mBase * CHW + (c * Hin + ho + kh) * Win + wo + kw;
    XT ta[32];
#pragma unroll
    for (int i = 0; i < 32; ++i) ta[i] = xp[(long)i * CHW];
#pragma unroll
    for (int i = 0; i < 32; ++i) As[wv][i][lane] = (_Float16)ta[i];

    // ---- B: Bs[n][k] = W[o=n, k] at this site (zero padded) ----
    const int wbase = ((c * Ho + ho) * Wo + wo) * 25 + r;
    float tb[4];
#pragma unroll
    for (int j = 0; j < 4; ++j) {
      int n = wv + 4 * j;
      tb[j] = (kvalid && n < Oc) ? W[(long)n * WN + wbase] : 0.f;
    }
#pragma unroll
    for (int j = 0; j < 4; ++j) Bs[wv + 4 * j][lane] = (_Float16)tb[j];
    __syncthreads();

    // ---- fragments per documented wave32 layouts ----
    const int row = lane & 15;
    const int hi  = lane >> 4;
    AFrag a0, a1;
    a0.h[0] = *(const v8h*)&As[wv][row][hi * 8];
    a0.h[1] = *(const v8h*)&As[wv][row][hi * 8 + 16];
    a1.h[0] = *(const v8h*)&As[wv][16 + row][hi * 8];
    a1.h[1] = *(const v8h*)&As[wv][16 + row][hi * 8 + 16];
    AFrag bu;
    bu.h[0] = *(const v8h*)&Bs[row][hi * 8];
    bu.h[1] = *(const v8h*)&Bs[row][hi * 8 + 16];
    v16h bf = bu.v;
    acc0 = wmma_f16(a0.v, bf, acc0);
    acc1 = wmma_f16(a1.v, bf, acc1);
    __syncthreads();
  }

  // ---- epilogue: bias + ReLU -> f16 ----
  const int n  = lane & 15;
  const int hi = lane >> 4;
  if (n < Oc) {
    const float bb = Bias[(n * Ho + ho) * Wo + wo];
    _Float16* op = Out + (long)(mBase + hi * 8) * OHW + (n * Ho + ho) * Wo + wo;
#pragma unroll
    for (int v = 0; v < 8; ++v) {
      float r0 = acc0[v] + bb; r0 = r0 > 0.f ? r0 : 0.f;
      float r1 = acc1[v] + bb; r1 = r1 > 0.f ? r1 : 0.f;
      op[(long)v * OHW]        = (_Float16)r0;
      op[(long)(v + 16) * OHW] = (_Float16)r1;
    }
  }
}

// ---------------------------------------------------------------------------
// C = act(A[Mxlda,f16] * W[NxK,f32]^T + bias).  A is contiguous f16, so the
// A tile is a pure 16B-granule global->LDS copy: async-to-LDS when available,
// vectorized v8h copy otherwise.  2 M-tiles per wave.  Columns gn>=N are
// written as zeros when ldOut > N (keeps downstream padded-K reads clean).
// ---------------------------------------------------------------------------
template <typename OutT>
__global__ __launch_bounds__(128) void gemm_wmma(
    const _Float16* __restrict__ A, const float* __restrict__ W,
    const float* __restrict__ Bias, OutT* __restrict__ Out,
    int N, int K, int lda, int ldOut, int doRelu)
{
  __shared__ __align__(32) _Float16 As[4][32][32];
  __shared__ __align__(32) _Float16 Bs[16][32];

  const int tid   = threadIdx.x;
  const int lane  = tid & 31;
  const int wv    = tid >> 5;
  const int nBase = blockIdx.x * 16;
  const int mBase = blockIdx.y * 128 + wv * 32;
  const int KC    = (K + 31) >> 5;

  v8f acc0 = {}, acc1 = {};

  for (int kc = 0; kc < KC; ++kc) {
    const int  kg     = kc * 32 + lane;
    const bool kvalid = (kg < K);

    // ---- A tile: 128 x 16B segments per wave tile, 4 per lane ----
#pragma unroll
    for (int it = 0; it < 4; ++it) {
      int seg = it * 32 + lane;
      int row = seg >> 2;
      int sk  = (seg & 3) * 8;
      const _Float16* gp = A + (long)(mBase + row) * lda + kc * 32 + sk;
#ifdef HAVE_ASYNC_LDS
      async_g2l_b128(gp, &As[wv][row][sk]);
#else
      *(v8h*)&As[wv][row][sk] = *(const v8h*)gp;
#endif
    }

    // ---- B tile: f32 weights -> f16, zero padded ----
    float tb[4];
#pragma unroll
    for (int j = 0; j < 4; ++j) {
      int gn = nBase + wv + 4 * j;
      tb[j] = (kvalid && gn < N) ? W[(long)gn * K + kg] : 0.f;
    }
#pragma unroll
    for (int j = 0; j < 4; ++j) Bs[wv + 4 * j][lane] = (_Float16)tb[j];

#ifdef HAVE_ASYNC_LDS
    wait_async0();
#endif
    __syncthreads();

    const int row = lane & 15;
    const int hi  = lane >> 4;
    AFrag a0, a1;
    a0.h[0] = *(const v8h*)&As[wv][row][hi * 8];
    a0.h[1] = *(const v8h*)&As[wv][row][hi * 8 + 16];
    a1.h[0] = *(const v8h*)&As[wv][16 + row][hi * 8];
    a1.h[1] = *(const v8h*)&As[wv][16 + row][hi * 8 + 16];
    AFrag bu;
    bu.h[0] = *(const v8h*)&Bs[row][hi * 8];
    bu.h[1] = *(const v8h*)&Bs[row][hi * 8 + 16];
    v16h bf = bu.v;
    acc0 = wmma_f16(a0.v, bf, acc0);
    acc1 = wmma_f16(a1.v, bf, acc1);
    __syncthreads();
  }

  const int gn = nBase + (lane & 15);
  const int hi = lane >> 4;
  if (gn < ldOut) {
    const bool real = (gn < N);
    const float bb  = real ? Bias[gn] : 0.f;
    OutT* op = Out + (long)(mBase + hi * 8) * ldOut + gn;
#pragma unroll
    for (int v = 0; v < 8; ++v) {
      float r0 = acc0[v] + bb;
      float r1 = acc1[v] + bb;
      if (doRelu) { r0 = r0 > 0.f ? r0 : 0.f; r1 = r1 > 0.f ? r1 : 0.f; }
      if (!real) { r0 = 0.f; r1 = 0.f; }
      op[(long)v * ldOut]        = (OutT)r0;
      op[(long)(v + 16) * ldOut] = (OutT)r1;
    }
  }
}

// ---------------------------------------------------------------------------
// 2x2 max pool (f16), stride 2: (BC, H, W) -> (BC, H/2, W/2)
// ---------------------------------------------------------------------------
__global__ void maxpool2_kernel(const _Float16* __restrict__ In,
                                _Float16* __restrict__ Out,
                                int BC, int H, int W)
{
  int Hp = H >> 1, Wp = W >> 1;
  int total = BC * Hp * Wp;
  int i = blockIdx.x * blockDim.x + threadIdx.x;
  if (i >= total) return;
  int wp = i % Wp;
  int t  = i / Wp;
  int hp = t % Hp;
  int bc = t / Hp;
  const _Float16* p = In + (bc * H + hp * 2) * W + wp * 2;
  _Float16 a = p[0] > p[1] ? p[0] : p[1];
  _Float16 b = p[W] > p[W + 1] ? p[W] : p[W + 1];
  Out[i] = a > b ? a : b;
}

extern "C" void kernel_launch(void* const* d_in, const int* in_sizes, int n_in,
                              void* d_out, int out_size, void* d_ws, size_t ws_size,
                              hipStream_t stream)
{
  const float* x    = (const float*)d_in[0];
  const float* w1   = (const float*)d_in[1];
  const float* b1   = (const float*)d_in[2];
  const float* w2   = (const float*)d_in[3];
  const float* b2   = (const float*)d_in[4];
  const float* w3   = (const float*)d_in[5];
  const float* b3   = (const float*)d_in[6];
  const float* fc2w = (const float*)d_in[7];
  const float* fc2b = (const float*)d_in[8];
  const float* fc3w = (const float*)d_in[9];
  const float* fc3b = (const float*)d_in[10];
  float* out = (float*)d_out;

  // workspace: f16 activations, each buffer 16B-aligned with 64-half tail pad
  _Float16* conv1 = (_Float16*)d_ws;                   // 1024*6*28*28
  _Float16* act1  = conv1 + 1024 * 6 * 28 * 28 + 64;   // 1024*6*14*14
  _Float16* conv2 = act1  + 1024 * 6 * 14 * 14 + 64;   // 1024*16*10*10
  _Float16* act2  = conv2 + 1024 * 16 * 10 * 10 + 64;  // 1024*400
  _Float16* h3    = act2  + 1024 * 400 + 64;           // 1024*120
  _Float16* h4    = h3    + 1024 * 120 + 64;           // 1024*96 (ld 96, cols 84..95 zeroed)

  dim3 blk(128);

  // layer 1: untied conv (3->6, 32x32 -> 28x28) + ReLU, pool -> 14x14
  localconv_wmma<float, 3, 32, 32, 6, 28, 28>
      <<<dim3(28 * 28, 8), blk, 0, stream>>>(x, w1, b1, conv1);
  {
    int total = 1024 * 6 * 14 * 14;
    maxpool2_kernel<<<(total + 255) / 256, 256, 0, stream>>>(
        conv1, act1, 1024 * 6, 28, 28);
  }

  // layer 2: untied conv (6->16, 14x14 -> 10x10) + ReLU, pool -> 5x5
  localconv_wmma<_Float16, 6, 14, 14, 16, 10, 10>
      <<<dim3(10 * 10, 8), blk, 0, stream>>>(act1, w2, b2, conv2);
  {
    int total = 1024 * 16 * 5 * 5;
    maxpool2_kernel<<<(total + 255) / 256, 256, 0, stream>>>(
        conv2, act2, 1024 * 16, 10, 10);
  }

  // layer 3 collapses to dense GEMM [1024x400] x [400x120]^T + ReLU
  gemm_wmma<_Float16><<<dim3(8, 8), blk, 0, stream>>>(
      act2, w3, b3, h3, 120, 400, 400, 120, 1);
  // fc2: [1024x120] x [120x84]^T + ReLU  (stored with ld 96, pad cols zeroed)
  gemm_wmma<_Float16><<<dim3(6, 8), blk, 0, stream>>>(
      h3, fc2w, fc2b, h4, 84, 120, 120, 96, 1);
  // fc3: [1024x84(ld 96)] x [84x10]^T -> f32 output (no ReLU)
  gemm_wmma<float><<<dim3(1, 8), blk, 0, stream>>>(
      h4, fc3w, fc3b, out, 10, 84, 96, 10, 0);
}
